// Attention_59803124630117
// MI455X (gfx1250) — matrix operand order repaired and checked by reading, not compiled, not run on hardware
//
#include <hip/hip_runtime.h>
#include <stdint.h>

// ---------------------------------------------------------------------------
// Types for CDNA5 WMMA (wave32): v_wmma_f32_16x16x32_bf16
// ---------------------------------------------------------------------------
typedef __attribute__((ext_vector_type(8)))  __bf16 v8bf;
typedef __attribute__((ext_vector_type(16))) __bf16 v16bf;
typedef __attribute__((ext_vector_type(8)))  float  v8f;

union AFrag { v16bf v; v8bf h[2]; };

#define NTOK  4096
#define CDIM  512
#define HEADS 8
#define DHEAD 64

static __device__ __forceinline__ float silu_f(float v) {
    return v / (1.0f + __expf(-v));
}

// ---------------------------------------------------------------------------
// Stage 0: f32 -> bf16 weight conversion (weights are reused many times in
// the GEMMs; pre-converting keeps hot loops to pure bf16 loads feeding wmma).
// ---------------------------------------------------------------------------
__global__ void cvt_bf16_kernel(const float* __restrict__ src,
                                __bf16* __restrict__ dst, int n) {
    int i = blockIdx.x * blockDim.x + threadIdx.x;
    if (i < n) dst[i] = (__bf16)src[i];
}

// ---------------------------------------------------------------------------
// Stage 1: xs = bf16( silu( LayerNorm(x) ) ),  one block per token row.
// ---------------------------------------------------------------------------
__global__ __launch_bounds__(256) void ln_silu_kernel(
        const float* __restrict__ x, __bf16* __restrict__ xs) {
    const int row = blockIdx.x;
    const int t   = threadIdx.x;                  // 256 threads, 2 cols each
    const float* xr = x + (size_t)row * CDIM;
    float a = xr[t], b = xr[t + 256];
    float s = a + b, ss = a * a + b * b;
    #pragma unroll
    for (int m = 16; m > 0; m >>= 1) { s += __shfl_xor(s, m); ss += __shfl_xor(ss, m); }
    __shared__ float red[2][8];
    const int w = t >> 5, ln = t & 31;
    if (ln == 0) { red[0][w] = s; red[1][w] = ss; }
    __syncthreads();
    s = 0.f; ss = 0.f;
    #pragma unroll
    for (int i = 0; i < 8; i++) { s += red[0][i]; ss += red[1][i]; }
    const float mean = s * (1.0f / CDIM);
    const float var  = ss * (1.0f / CDIM) - mean * mean;
    const float rs   = rsqrtf(var + 1e-5f);
    float ya = (a - mean) * rs, yb = (b - mean) * rs;
    xs[(size_t)row * CDIM + t]       = (__bf16)silu_f(ya);
    xs[(size_t)row * CDIM + t + 256] = (__bf16)silu_f(yb);
}

// ---------------------------------------------------------------------------
// Stage 2: q/k/v = xs @ W + b, WMMA GEMM. grid = (M/32, N/64, 3).
// One wave per block computes a 32x64 f32 tile (2x4 accumulator blocks).
// ---------------------------------------------------------------------------
__global__ __launch_bounds__(32) void qkv_gemm_kernel(
        const __bf16* __restrict__ xs, const __bf16* __restrict__ wbf,
        const float* __restrict__ bq, const float* __restrict__ bk,
        const float* __restrict__ bv, float* __restrict__ qkv) {
    const int z = blockIdx.z;
    const __bf16* W    = wbf + (size_t)z * CDIM * CDIM;
    const float*  bias = (z == 0) ? bq : ((z == 1) ? bk : bv);
    float*        out  = qkv + (size_t)z * NTOK * CDIM;

    const int lane = threadIdx.x & 31;
    const int half = lane >> 4, ln = lane & 15;
    const int m0 = blockIdx.x * 32, n0 = blockIdx.y * 64;

    v8f acc[2][4] = {};
    for (int kk = 0; kk < CDIM; kk += 32) {
        AFrag a[2];
        #pragma unroll
        for (int i = 0; i < 2; i++) {
            const __bf16* p = xs + (size_t)(m0 + 16 * i + ln) * CDIM + kk + 8 * half;
            a[i].h[0] = *(const v8bf*)p;
            a[i].h[1] = *(const v8bf*)(p + 16);
        }
        #pragma unroll
        for (int b = 0; b < 4; b++) {
            v16bf bf;
            #pragma unroll
            for (int e = 0; e < 8; e++) {
                bf[e]     = W[(size_t)(kk + 8 * half + e) * CDIM + n0 + 16 * b + ln];
                bf[8 + e] = W[(size_t)(kk + 16 + 8 * half + e) * CDIM + n0 + 16 * b + ln];
            }
            #pragma unroll
            for (int i = 0; i < 2; i++)
                acc[i][b] = __builtin_amdgcn_wmma_f32_16x16x32_bf16(
                    false, a[i].v, false, bf, (short)0, acc[i][b], false, false);
        }
    }
    #pragma unroll
    for (int i = 0; i < 2; i++) {
        #pragma unroll
        for (int b = 0; b < 4; b++) {
            const int col = n0 + 16 * b + ln;
            const float bs = bias[col];
            #pragma unroll
            for (int r = 0; r < 8; r++) {
                const int row = m0 + 16 * i + r + 8 * half;
                out[(size_t)row * CDIM + col] = acc[i][b][r] + bs;
            }
        }
    }
}

// ---------------------------------------------------------------------------
// Stage 3: LN(q)*g+b (scaled by inner^-0.5), LN(k)*g+b, repack to head
// layouts:  qh,vh = [H][N][64] bf16 (row major),  kt = [H][64][N] bf16
// (pre-transposed so each 32-key K^T tile row is a contiguous 64B copy).
// ---------------------------------------------------------------------------
__global__ __launch_bounds__(256) void qkln_kernel(
        const float* __restrict__ qkv,
        const float* __restrict__ gq, const float* __restrict__ beq,
        const float* __restrict__ gk, const float* __restrict__ bek,
        __bf16* __restrict__ qh, __bf16* __restrict__ kt, __bf16* __restrict__ vh) {
    const int n = blockIdx.x;
    const int t = threadIdx.x;
    const float* q = qkv + (size_t)n * CDIM;
    const float* k = qkv + (size_t)(NTOK + n) * CDIM;
    const float* v = qkv + (size_t)(2 * NTOK + n) * CDIM;

    float q0 = q[t], q1 = q[t + 256], k0 = k[t], k1 = k[t + 256];
    float sq = q0 + q1, sqq = q0 * q0 + q1 * q1;
    float sk = k0 + k1, skk = k0 * k0 + k1 * k1;
    #pragma unroll
    for (int m = 16; m > 0; m >>= 1) {
        sq += __shfl_xor(sq, m); sqq += __shfl_xor(sqq, m);
        sk += __shfl_xor(sk, m); skk += __shfl_xor(skk, m);
    }
    __shared__ float red[4][8];
    const int w = t >> 5, ln = t & 31;
    if (ln == 0) { red[0][w] = sq; red[1][w] = sqq; red[2][w] = sk; red[3][w] = skk; }
    __syncthreads();
    sq = sqq = sk = skk = 0.f;
    #pragma unroll
    for (int i = 0; i < 8; i++) {
        sq += red[0][i]; sqq += red[1][i]; sk += red[2][i]; skk += red[3][i];
    }
    const float mq = sq * (1.0f / CDIM);
    const float rq = rsqrtf(sqq * (1.0f / CDIM) - mq * mq + 1e-5f);
    const float mk = sk * (1.0f / CDIM);
    const float rk = rsqrtf(skk * (1.0f / CDIM) - mk * mk + 1e-5f);
    const float qscale = 0.04419417382415922f;    // 512^-0.5 (inner, not dhead)

    #pragma unroll
    for (int p = 0; p < 2; p++) {
        const int c = t + 256 * p;
        const float qv = (p == 0) ? q0 : q1;
        const float kv = (p == 0) ? k0 : k1;
        const float yq = (((qv - mq) * rq) * gq[c] + beq[c]) * qscale;
        const float yk = ((kv - mk) * rk) * gk[c] + bek[c];
        const int h = c >> 6, dd = c & 63;
        qh[((size_t)h * NTOK + n) * DHEAD + dd] = (__bf16)yq;
        kt[((size_t)h * DHEAD + dd) * NTOK + n] = (__bf16)yk;
        vh[((size_t)h * NTOK + n) * DHEAD + dd] = (__bf16)v[c];
    }
}

// ---------------------------------------------------------------------------
// Stage 4: flash attention. grid = (H, N/64), block = 128 (4 waves).
// Each wave: 16 query rows, online softmax over key tiles of 32.
// K^T/V tiles are block-uniform, so they are staged into double-buffered LDS
// with GLOBAL_LOAD_ASYNC_TO_LDS_B128 (ASYNCcnt) issued one tile ahead; the
// copy overlaps the WMMA + softmax work on the current tile. P is re-shaped
// from C-layout to A-layout through a per-wave LDS staging buffer.
// ---------------------------------------------------------------------------
__global__ __launch_bounds__(128) void attn_kernel(
        const __bf16* __restrict__ qh, const __bf16* __restrict__ kt,
        const __bf16* __restrict__ vh, float* __restrict__ ob) {
    const int h     = blockIdx.x;
    const int q0    = blockIdx.y * 64;
    const int t     = threadIdx.x;
    const int wave  = t >> 5;
    const int lane  = t & 31;
    const int half  = lane >> 4, ln = lane & 15;
    const int qbase = q0 + wave * 16;

    const __bf16* Q  = qh + (size_t)h * NTOK * DHEAD;
    const __bf16* KT = kt + (size_t)h * DHEAD * NTOK;
    const __bf16* V  = vh + (size_t)h * NTOK * DHEAD;

    __shared__ __align__(32) __bf16 pst[4][16 * 32];         // per-wave P staging
    __shared__ __align__(32) __bf16 kbuf[2][DHEAD * 32];     // K^T tile [64][32]
    __shared__ __align__(32) __bf16 vbuf[2][32 * DHEAD];     // V tile  [32][64]
    __bf16* ps = pst[wave];

    // Cooperative async copy of the K^T and V tiles for key block j0 -> buf b.
    // 128 threads x 16B x 2 chunks per tile (4KB each), tracked on ASYNCcnt.
    auto stage = [&](int j0, int b) {
        {   // K^T tile: 64 rows x 32 cols bf16 (64B per row = 4 x b128)
            const int c = t & 3, r0 = t >> 2;                // chunk, base row
            #pragma unroll
            for (int rr = 0; rr < 2; rr++) {
                const int d = r0 + 32 * rr;
                uint64_t ga  = (uint64_t)(uintptr_t)(KT + (size_t)d * NTOK + j0 + 8 * c);
                uint32_t dst = (uint32_t)(uintptr_t)&kbuf[b][d * 32 + 8 * c];
                asm volatile("global_load_async_to_lds_b128 %0, %1, off"
                             :: "v"(dst), "v"(ga) : "memory");
            }
        }
        {   // V tile: 32 rows x 64 cols bf16 (128B per row = 8 x b128)
            const int c = t & 7, r0 = t >> 3;
            #pragma unroll
            for (int rr = 0; rr < 2; rr++) {
                const int r = r0 + 16 * rr;
                uint64_t ga  = (uint64_t)(uintptr_t)(V + (size_t)(j0 + r) * DHEAD + 8 * c);
                uint32_t dst = (uint32_t)(uintptr_t)&vbuf[b][r * DHEAD + 8 * c];
                asm volatile("global_load_async_to_lds_b128 %0, %1, off"
                             :: "v"(dst), "v"(ga) : "memory");
            }
        }
    };

    // Q fragments held in registers for the whole key loop (d = 64 -> 2 frags)
    AFrag qf[2];
    #pragma unroll
    for (int kkk = 0; kkk < 2; kkk++) {
        const __bf16* p = Q + (size_t)(qbase + ln) * DHEAD + 32 * kkk + 8 * half;
        qf[kkk].h[0] = *(const v8bf*)p;
        qf[kkk].h[1] = *(const v8bf*)(p + 16);
    }

    v8f o[4] = {};
    float mstate[8], lstate[8];
    #pragma unroll
    for (int r = 0; r < 8; r++) { mstate[r] = -1e30f; lstate[r] = 0.f; }

    stage(0, 0);                                   // prime the pipeline
    for (int j0 = 0; j0 < NTOK; j0 += 32) {
        const int buf = (j0 >> 5) & 1;
        asm volatile("s_wait_asynccnt 0x0" ::: "memory");   // our copies landed
        __syncthreads();        // all copies landed; prev readers of buf^1 done
        if (j0 + 32 < NTOK) stage(j0 + 32, buf ^ 1);

        // ---- S = Q @ K^T  (16 x 32 tile, two 16-col blocks) from LDS ----
        v8f s[2] = {};
        #pragma unroll
        for (int c = 0; c < 2; c++) {
            #pragma unroll
            for (int kkk = 0; kkk < 2; kkk++) {
                v16bf bf;
                #pragma unroll
                for (int e = 0; e < 8; e++) {
                    bf[e]     = kbuf[buf][(32 * kkk + 8 * half + e) * 32 + 16 * c + ln];
                    bf[8 + e] = kbuf[buf][(32 * kkk + 16 + 8 * half + e) * 32 + 16 * c + ln];
                }
                s[c] = __builtin_amdgcn_wmma_f32_16x16x32_bf16(
                    false, qf[kkk].v, false, bf, (short)0, s[c], false, false);
            }
        }
        // ---- online softmax: rows 0-7 live in lanes 0-15, 8-15 in 16-31 ----
        #pragma unroll
        for (int r = 0; r < 8; r++) {
            float mx = fmaxf(s[0][r], s[1][r]);
            mx = fmaxf(mx, __shfl_xor(mx, 1)); mx = fmaxf(mx, __shfl_xor(mx, 2));
            mx = fmaxf(mx, __shfl_xor(mx, 4)); mx = fmaxf(mx, __shfl_xor(mx, 8));
            const float mnew  = fmaxf(mstate[r], mx);
            const float alpha = __expf(mstate[r] - mnew);
            const float p0 = __expf(s[0][r] - mnew);
            const float p1 = __expf(s[1][r] - mnew);
            float rsum = p0 + p1;
            rsum += __shfl_xor(rsum, 1); rsum += __shfl_xor(rsum, 2);
            rsum += __shfl_xor(rsum, 4); rsum += __shfl_xor(rsum, 8);
            lstate[r] = lstate[r] * alpha + rsum;
            mstate[r] = mnew;
            #pragma unroll
            for (int b = 0; b < 4; b++) o[b][r] *= alpha;     // rescale O row
            const int row = r + 8 * half;                      // stage P (bf16)
            ps[row * 32 + ln]      = (__bf16)p0;
            ps[row * 32 + 16 + ln] = (__bf16)p1;
        }
        asm volatile("s_wait_dscnt 0x0" ::: "memory");  // LDS turn-around
        // ---- P as A-fragment from LDS ----
        AFrag pf;
        {
            const __bf16* p = ps + ln * 32 + 8 * half;
            pf.h[0] = *(const v8bf*)p;
            pf.h[1] = *(const v8bf*)(p + 16);
        }
        // ---- O += P @ V (V from LDS) ----
        #pragma unroll
        for (int b = 0; b < 4; b++) {
            v16bf vb;
            #pragma unroll
            for (int e = 0; e < 8; e++) {
                vb[e]     = vbuf[buf][(8 * half + e) * DHEAD + 16 * b + ln];
                vb[8 + e] = vbuf[buf][(16 + 8 * half + e) * DHEAD + 16 * b + ln];
            }
            o[b] = __builtin_amdgcn_wmma_f32_16x16x32_bf16(
                false, pf.v, false, vb, (short)0, o[b], false, false);
        }
    }
    // ---- normalize and write O back as [N][inner] f32 ----
    #pragma unroll
    for (int r = 0; r < 8; r++) {
        const float inv = 1.0f / lstate[r];
        const int row = qbase + r + 8 * half;
        #pragma unroll
        for (int b = 0; b < 4; b++)
            ob[(size_t)row * CDIM + h * DHEAD + 16 * b + ln] = o[b][r] * inv;
    }
}

// ---------------------------------------------------------------------------
// Stage 5: out = silu(O) @ w_o + b_o. SiLU fused into the A-fragment loader.
// ---------------------------------------------------------------------------
__global__ __launch_bounds__(32) void out_gemm_kernel(
        const float* __restrict__ ob, const __bf16* __restrict__ wo,
        const float* __restrict__ bo, float* __restrict__ out) {
    const int lane = threadIdx.x & 31;
    const int half = lane >> 4, ln = lane & 15;
    const int m0 = blockIdx.x * 32, n0 = blockIdx.y * 64;

    v8f acc[2][4] = {};
    for (int kk = 0; kk < CDIM; kk += 32) {
        AFrag a[2];
        #pragma unroll
        for (int i = 0; i < 2; i++) {
            const float* p = ob + (size_t)(m0 + 16 * i + ln) * CDIM + kk + 8 * half;
            #pragma unroll
            for (int e = 0; e < 8; e++) a[i].h[0][e] = (__bf16)silu_f(p[e]);
            #pragma unroll
            for (int e = 0; e < 8; e++) a[i].h[1][e] = (__bf16)silu_f(p[16 + e]);
        }
        #pragma unroll
        for (int b = 0; b < 4; b++) {
            v16bf bf;
            #pragma unroll
            for (int e = 0; e < 8; e++) {
                bf[e]     = wo[(size_t)(kk + 8 * half + e) * CDIM + n0 + 16 * b + ln];
                bf[8 + e] = wo[(size_t)(kk + 16 + 8 * half + e) * CDIM + n0 + 16 * b + ln];
            }
            #pragma unroll
            for (int i = 0; i < 2; i++)
                acc[i][b] = __builtin_amdgcn_wmma_f32_16x16x32_bf16(
                    false, a[i].v, false, bf, (short)0, acc[i][b], false, false);
        }
    }
    #pragma unroll
    for (int i = 0; i < 2; i++) {
        #pragma unroll
        for (int b = 0; b < 4; b++) {
            const int col = n0 + 16 * b + ln;
            const float bs = bo[col];
            #pragma unroll
            for (int r = 0; r < 8; r++) {
                const int row = m0 + 16 * i + r + 8 * half;
                out[(size_t)row * CDIM + col] = acc[i][b][r] + bs;
            }
        }
    }
}

// ---------------------------------------------------------------------------
// Launch: all on `stream`, all scratch carved from d_ws (~50 MiB).
// ---------------------------------------------------------------------------
extern "C" void kernel_launch(void* const* d_in, const int* in_sizes, int n_in,
                              void* d_out, int out_size, void* d_ws, size_t ws_size,
                              hipStream_t stream) {
    const float* x    = (const float*)d_in[0];
    const float* w_q  = (const float*)d_in[1];
    const float* b_q  = (const float*)d_in[2];
    const float* w_k  = (const float*)d_in[3];
    const float* b_k  = (const float*)d_in[4];
    const float* w_v  = (const float*)d_in[5];
    const float* b_v  = (const float*)d_in[6];
    const float* g_q  = (const float*)d_in[7];
    const float* be_q = (const float*)d_in[8];
    const float* g_k  = (const float*)d_in[9];
    const float* be_k = (const float*)d_in[10];
    const float* w_o  = (const float*)d_in[11];
    const float* b_o  = (const float*)d_in[12];
    float* out = (float*)d_out;

    char* ws = (char*)d_ws;
    const size_t MB = 1024 * 1024;
    __bf16* wbf = (__bf16*)(ws);                 // [0,2) MiB   : 4 weight mats bf16
    __bf16* xs  = (__bf16*)(ws + 2 * MB);        // [2,6) MiB   : silu(LN(x)) bf16
    float*  qkv = (float*)(ws + 6 * MB);         // [6,30) MiB  : q,k,v f32
    __bf16* qh  = (__bf16*)(ws + 30 * MB);       // [30,34) MiB : q heads bf16
    __bf16* kt  = (__bf16*)(ws + 34 * MB);       // [34,38) MiB : k^T heads bf16
    __bf16* vh  = (__bf16*)(ws + 38 * MB);       // [38,42) MiB : v heads bf16
    float*  ob  = (float*)(ws + 42 * MB);        // [42,50) MiB : attn out f32

    const int WN = CDIM * CDIM;                  // 262144 per matrix
    cvt_bf16_kernel<<<WN / 256, 256, 0, stream>>>(w_q, wbf + 0 * (size_t)WN, WN);
    cvt_bf16_kernel<<<WN / 256, 256, 0, stream>>>(w_k, wbf + 1 * (size_t)WN, WN);
    cvt_bf16_kernel<<<WN / 256, 256, 0, stream>>>(w_v, wbf + 2 * (size_t)WN, WN);
    cvt_bf16_kernel<<<WN / 256, 256, 0, stream>>>(w_o, wbf + 3 * (size_t)WN, WN);

    ln_silu_kernel<<<NTOK, 256, 0, stream>>>(x, xs);

    qkv_gemm_kernel<<<dim3(NTOK / 32, CDIM / 64, 3), 32, 0, stream>>>(
        xs, wbf, b_q, b_k, b_v, qkv);

    qkln_kernel<<<NTOK, 256, 0, stream>>>(qkv, g_q, be_q, g_k, be_k, qh, kt, vh);

    attn_kernel<<<dim3(HEADS, NTOK / 64), 128, 0, stream>>>(qh, kt, vh, ob);

    out_gemm_kernel<<<dim3(NTOK / 32, CDIM / 64), 32, 0, stream>>>(
        ob, wbf + 3 * (size_t)WN, b_o, out);
}
